// DifferentialMultiHeadAttention_3324304687175
// MI455X (gfx1250) — matrix operand order repaired and checked by reading, not compiled, not run on hardware
//
#include <hip/hip_runtime.h>
#include <hip/hip_bf16.h>

// Problem constants
#define Bb   4
#define Nn   1024
#define Dd   512
#define Hh   8
#define DH   64
#define LAM  0.5f

typedef __attribute__((ext_vector_type(16))) _Float16 v16h;
typedef __attribute__((ext_vector_type(8)))  _Float16 v8h;
typedef __attribute__((ext_vector_type(8)))  float    v8f;

// ---------------------------------------------------------------------------
// Fragment loaders matching CDNA5 WMMA 16-bit VGPR layouts (CDNA5 ISA)
// ---------------------------------------------------------------------------
__device__ __forceinline__ v16h frag_a(const _Float16* base, int ld) {
  const int l  = threadIdx.x & 31;
  const _Float16* p = base + (size_t)(l & 15) * ld + ((l & 16) ? 8 : 0);
  v8h lo = *(const v8h*)p;
  v8h hi = *(const v8h*)(p + 16);
  v16h r;
#pragma unroll
  for (int i = 0; i < 8; ++i) { r[i] = lo[i]; r[i + 8] = hi[i]; }
  return r;
}

__device__ __forceinline__ v16h frag_b(const _Float16* base, int ld) {
  const int l = threadIdx.x & 31;
  const _Float16* p = base + (size_t)(l & 15) * ld + ((l & 16) ? 8 : 0);
  v8h lo = *(const v8h*)p;
  v8h hi = *(const v8h*)(p + 16);
  v16h r;
#pragma unroll
  for (int i = 0; i < 8; ++i) { r[i] = lo[i]; r[i + 8] = hi[i]; }
  return r;
}

__device__ __forceinline__ v8f wmma16(v16h a, v16h b, v8f c) {
  return __builtin_amdgcn_wmma_f32_16x16x32_f16(false, a, false, b,
                                                (short)0, c, false, false);
}

// Async copy of 16 bytes: global memory -> LDS, tracked by ASYNCcnt.
__device__ __forceinline__ void async_copy_b128(void* lds_dst,
                                                const void* gsrc) {
  unsigned lds_off = (unsigned)(uintptr_t)lds_dst;
  unsigned long long gaddr = (unsigned long long)(uintptr_t)gsrc;
  asm volatile("global_load_async_to_lds_b128 %0, %1, off"
               :: "v"(lds_off), "v"(gaddr)
               : "memory");
}
__device__ __forceinline__ void wait_async_all() {
  asm volatile("s_wait_asynccnt 0x0" ::: "memory");
}

// ---------------------------------------------------------------------------
// Cross-lane XOR butterflies within 16-lane rows via v_permlane16_b32
// (pure VALU; rows [0,16) and [16,32) permute independently -> our per-half
// row semantics). Selector nibble for lane i is i^m.
// ---------------------------------------------------------------------------
#if __has_builtin(__builtin_amdgcn_permlane16)
template <unsigned S0, unsigned S1>
__device__ __forceinline__ float lane_perm16(float v) {
  int vi = __float_as_int(v);
  return __int_as_float(
      __builtin_amdgcn_permlane16(vi, vi, (int)S0, (int)S1, false, false));
}
__device__ __forceinline__ float xor1(float v) { return lane_perm16<0x67452301u, 0xEFCDAB89u>(v); }
__device__ __forceinline__ float xor2(float v) { return lane_perm16<0x54761032u, 0xDCFE98BAu>(v); }
__device__ __forceinline__ float xor4(float v) { return lane_perm16<0x32107654u, 0xBA98FEDCu>(v); }
__device__ __forceinline__ float xor8(float v) { return lane_perm16<0xFEDCBA98u, 0x76543210u>(v); }
#else
__device__ __forceinline__ float xor1(float v) { return __shfl_xor(v, 1, 32); }
__device__ __forceinline__ float xor2(float v) { return __shfl_xor(v, 2, 32); }
__device__ __forceinline__ float xor4(float v) { return __shfl_xor(v, 4, 32); }
__device__ __forceinline__ float xor8(float v) { return __shfl_xor(v, 8, 32); }
#endif

__device__ __forceinline__ float redmax16(float v) {
  v = fmaxf(v, xor1(v));
  v = fmaxf(v, xor2(v));
  v = fmaxf(v, xor4(v));
  v = fmaxf(v, xor8(v));
  return v;
}
__device__ __forceinline__ float redsum16(float v) {
  v += xor1(v);
  v += xor2(v);
  v += xor4(v);
  v += xor8(v);
  return v;
}

// ---------------------------------------------------------------------------
// fp32 -> f16 convert (grid-stride)
// ---------------------------------------------------------------------------
__global__ void cvt_f16_kernel(const float* __restrict__ in,
                               _Float16* __restrict__ out, int n) {
  for (int i = blockIdx.x * blockDim.x + threadIdx.x; i < n;
       i += gridDim.x * blockDim.x)
    out[i] = (_Float16)in[i];
}

// W (K x Nc, row-major) -> Wt (Nc x K, row-major) in f16
__global__ void transpose_w_kernel(const float* __restrict__ W,
                                   _Float16* __restrict__ Wt, int K, int Nc) {
  int total = K * Nc;
  for (int i = blockIdx.x * blockDim.x + threadIdx.x; i < total;
       i += gridDim.x * blockDim.x) {
    int n = i / K, k = i - n * K;
    Wt[(size_t)n * K + k] = (_Float16)W[(size_t)k * Nc + n];
  }
}

// ---------------------------------------------------------------------------
// Projection GEMM: C(M=4096 x Nc) = A(f16, M x 512) @ Bt(f16, Nc x 512)^T + bias
// MODE 0: store f16 row-major (M x Nc)         -> Q / K
// MODE 1: store f16 as Vt (B, H, DH, N) layout -> V transposed for attention
// ---------------------------------------------------------------------------
template <int MODE>
__global__ __launch_bounds__(256) void proj_gemm_kernel(
    const _Float16* __restrict__ A, const _Float16* __restrict__ Bt,
    const float* __restrict__ bias, _Float16* __restrict__ C, int Nc) {
  const int K = Dd;  // 512
  const int w = threadIdx.x >> 5, lane = threadIdx.x & 31;
  const int wm = w >> 2, wn = w & 3;
  const int row0 = blockIdx.y * 128 + wm * 64;
  const int col0 = blockIdx.x * 128 + wn * 32;
  const v8f vz = {0.f, 0.f, 0.f, 0.f, 0.f, 0.f, 0.f, 0.f};

  v8f acc[4][2];
#pragma unroll
  for (int mt = 0; mt < 4; ++mt)
#pragma unroll
    for (int nt = 0; nt < 2; ++nt) acc[mt][nt] = vz;

  for (int kk = 0; kk < K; kk += 32) {
    v16h a[4];
#pragma unroll
    for (int mt = 0; mt < 4; ++mt)
      a[mt] = frag_a(A + (size_t)(row0 + mt * 16) * K + kk, K);
#pragma unroll
    for (int nt = 0; nt < 2; ++nt) {
      v16h bf = frag_b(Bt + (size_t)(col0 + nt * 16) * K + kk, K);
#pragma unroll
      for (int mt = 0; mt < 4; ++mt)
        acc[mt][nt] = wmma16(a[mt], bf, acc[mt][nt]);
    }
  }

  const int hi8 = (lane & 16) ? 8 : 0;
#pragma unroll
  for (int mt = 0; mt < 4; ++mt) {
#pragma unroll
    for (int nt = 0; nt < 2; ++nt) {
      const int n = col0 + nt * 16 + (lane & 15);
      const float bv = bias[n];
#pragma unroll
      for (int e = 0; e < 8; ++e) {
        const int m = row0 + mt * 16 + e + hi8;
        const float val = acc[mt][nt][e] + bv;
        if (MODE == 0) {
          C[(size_t)m * Nc + n] = (_Float16)val;
        } else {  // Vt: [b][h][dh][n]
          const int bb = m >> 10, nnq = m & 1023;
          const int hh = n >> 6, dh = n & 63;
          C[(((size_t)(bb * Hh + hh) * DH + dh) << 10) + nnq] = (_Float16)val;
        }
      }
    }
  }
}

// ---------------------------------------------------------------------------
// Differential attention. Grid: (N/128 query blocks, B*H). 256 threads.
// Each wave owns 16 query rows; online softmax per 64-key block (one
// max/sum reduction and one O-rescale per row per block), both branches,
// out = O1/z1 - LAM * O2/z2 in fp32. K/V tiles staged with
// GLOBAL_LOAD_ASYNC_TO_LDS_B128 (ASYNCcnt).
// ---------------------------------------------------------------------------
__global__ __launch_bounds__(256) void diff_attn_kernel(
    const _Float16* __restrict__ Qh, const _Float16* __restrict__ Kh,
    const _Float16* __restrict__ Vt, float* __restrict__ Out) {
  __shared__ _Float16 sK[64 * 128];      // [key][k1|k2]      16 KB
  __shared__ _Float16 sV[64 * 64];       // [dh][key]          8 KB
  __shared__ _Float16 sP[8][16 * 64];    // per-wave P stage  16 KB

  const int tid = threadIdx.x;
  const int w = tid >> 5, lane = tid & 31;
  const int lanelo = lane & 15;
  const int hi8 = (lane & 16) ? 8 : 0;
  const int bh = blockIdx.y;
  const int b = bh >> 3, h = bh & 7;
  const int q0 = blockIdx.x * 128;
  const v8f vz = {0.f, 0.f, 0.f, 0.f, 0.f, 0.f, 0.f, 0.f};

  // Q fragments for this wave's 16 rows, both branches, K split 2x32.
  const _Float16* qbase =
      Qh + (size_t)(b * Nn + q0 + w * 16) * (2 * Dd) + h * 128;
  v16h aq[2][2];
#pragma unroll
  for (int pass = 0; pass < 2; ++pass)
#pragma unroll
    for (int j = 0; j < 2; ++j)
      aq[pass][j] = frag_a(qbase + pass * 64 + j * 32, 2 * Dd);

  v8f O[2][4];
  float mstat[2][8], zstat[2][8];
#pragma unroll
  for (int p = 0; p < 2; ++p) {
#pragma unroll
    for (int t = 0; t < 4; ++t) O[p][t] = vz;
#pragma unroll
    for (int e = 0; e < 8; ++e) { mstat[p][e] = -1e30f; zstat[p][e] = 0.f; }
  }

  for (int kb = 0; kb < Nn / 64; ++kb) {
    const int key0 = kb * 64;
    // Async stage: K block (64 keys x 128) and Vt block (64 dh x 64 keys).
#pragma unroll
    for (int i = 0; i < 4; ++i) {
      int idx8 = tid + i * 256;
      int row = idx8 >> 4, col = (idx8 & 15) * 8;
      async_copy_b128(
          &sK[row * 128 + col],
          &Kh[(size_t)(b * Nn + key0 + row) * (2 * Dd) + h * 128 + col]);
    }
#pragma unroll
    for (int i = 0; i < 2; ++i) {
      int idx8 = tid + i * 256;
      int dh = idx8 >> 3, key = (idx8 & 7) * 8;
      async_copy_b128(
          &sV[dh * 64 + key],
          &Vt[((size_t)((b * Hh + h) * DH + dh) << 10) + key0 + key]);
    }
    if (kb + 1 < Nn / 64)  // global_prefetch_b8 of next K block
      __builtin_prefetch(&Kh[(size_t)(b * Nn + key0 + 64) * (2 * Dd) + h * 128],
                         0, 0);
    wait_async_all();   // s_wait_asynccnt 0: this wave's copies landed in LDS
    __syncthreads();    // all waves' copies landed

    _Float16* sPw = sP[w];
#pragma unroll
    for (int pass = 0; pass < 2; ++pass) {
      const int koff = pass * 64;
      // All four 16-key S tiles first (8 WMMAs), then one softmax update.
      v8f st[4];
#pragma unroll
      for (int nt = 0; nt < 4; ++nt) {
        st[nt] = vz;
#pragma unroll
        for (int j = 0; j < 2; ++j)
          st[nt] = wmma16(aq[pass][j],
                          frag_b(&sK[(nt * 16) * 128 + koff + j * 32], 128),
                          st[nt]);
      }
#pragma unroll
      for (int e = 0; e < 8; ++e) {
        const float s0 = st[0][e] * 0.125f;  // 1/sqrt(DH)
        const float s1 = st[1][e] * 0.125f;
        const float s2 = st[2][e] * 0.125f;
        const float s3 = st[3][e] * 0.125f;
        const float rm = redmax16(fmaxf(fmaxf(s0, s1), fmaxf(s2, s3)));
        const float mold = mstat[pass][e];
        const float mnew = fmaxf(mold, rm);
        const float corr = __expf(mold - mnew);
        const float p0 = __expf(s0 - mnew);
        const float p1 = __expf(s1 - mnew);
        const float p2 = __expf(s2 - mnew);
        const float p3 = __expf(s3 - mnew);
        zstat[pass][e] =
            zstat[pass][e] * corr + redsum16((p0 + p1) + (p2 + p3));
        mstat[pass][e] = mnew;
#pragma unroll
        for (int t = 0; t < 4; ++t) O[pass][t][e] *= corr;
        _Float16* prow = &sPw[(e + hi8) * 64 + lanelo];
        prow[0]  = (_Float16)p0;
        prow[16] = (_Float16)p1;
        prow[32] = (_Float16)p2;
        prow[48] = (_Float16)p3;
      }
      asm volatile("s_wait_dscnt 0" ::: "memory");  // P staged -> readable
      v16h aP[2];
#pragma unroll
      for (int j = 0; j < 2; ++j) aP[j] = frag_a(&sPw[j * 32], 64);
#pragma unroll
      for (int dt = 0; dt < 4; ++dt)
#pragma unroll
        for (int j = 0; j < 2; ++j)
          O[pass][dt] = wmma16(aP[j],
                               frag_b(&sV[(dt * 16) * 64 + j * 32], 64),
                               O[pass][dt]);
    }
    __syncthreads();
  }

  // out = O1/z1 - LAM * O2/z2, fp32, layout (B, N, D) with col = h*64 + dh
#pragma unroll
  for (int dt = 0; dt < 4; ++dt) {
#pragma unroll
    for (int e = 0; e < 8; ++e) {
      const int n = q0 + w * 16 + e + hi8;
      const int col = h * 64 + dt * 16 + lanelo;
      const float val =
          O[0][dt][e] / zstat[0][e] - LAM * (O[1][dt][e] / zstat[1][e]);
      Out[(size_t)(b * Nn + n) * Dd + col] = val;
    }
  }
}

// ---------------------------------------------------------------------------
extern "C" void kernel_launch(void* const* d_in, const int* in_sizes, int n_in,
                              void* d_out, int out_size, void* d_ws,
                              size_t ws_size, hipStream_t stream) {
  const float* x  = (const float*)d_in[0];
  const float* Wq = (const float*)d_in[1];
  const float* bq = (const float*)d_in[2];
  const float* Wk = (const float*)d_in[3];
  const float* bk = (const float*)d_in[4];
  const float* Wv = (const float*)d_in[5];
  const float* bv = (const float*)d_in[6];
  float* out = (float*)d_out;

  char* ws = (char*)d_ws;
  _Float16* xh  = (_Float16*)(ws);                           //  4 MB: x f16
  _Float16* Wqt = (_Float16*)(ws + (size_t)4 * 1024 * 1024); //  1 MB
  _Float16* Wkt = (_Float16*)(ws + (size_t)5 * 1024 * 1024); //  1 MB
  _Float16* Wvt = (_Float16*)(ws + (size_t)6 * 1024 * 1024); // .5 MB
  _Float16* Qh  = (_Float16*)(ws + (size_t)6 * 1024 * 1024 + 512 * 1024); // 8 MB
  _Float16* Kh  = Qh + (size_t)Bb * Nn * 2 * Dd;             //  8 MB
  _Float16* Vt  = Kh + (size_t)Bb * Nn * 2 * Dd;             //  4 MB

  const int xelems = Bb * Nn * Dd;  // 2M
  cvt_f16_kernel<<<2048, 256, 0, stream>>>(x, xh, xelems);
  transpose_w_kernel<<<2048, 256, 0, stream>>>(Wq, Wqt, Dd, 2 * Dd);
  transpose_w_kernel<<<2048, 256, 0, stream>>>(Wk, Wkt, Dd, 2 * Dd);
  transpose_w_kernel<<<1024, 256, 0, stream>>>(Wv, Wvt, Dd, Dd);

  // M = B*N = 4096 rows -> grid.y = 32 tiles of 128
  proj_gemm_kernel<0><<<dim3(8, 32), 256, 0, stream>>>(xh, Wqt, bq, Qh, 2 * Dd);
  proj_gemm_kernel<0><<<dim3(8, 32), 256, 0, stream>>>(xh, Wkt, bk, Kh, 2 * Dd);
  proj_gemm_kernel<1><<<dim3(4, 32), 256, 0, stream>>>(xh, Wvt, bv, Vt, Dd);

  diff_attn_kernel<<<dim3(Nn / 128, Bb * Hh), 256, 0, stream>>>(Qh, Kh, Vt, out);
}
